// DBlock_36584531427335
// MI455X (gfx1250) — matrix operand order repaired and checked by reading, not compiled, not run on hardware
//
#include <hip/hip_runtime.h>
#include <hip/hip_bf16.h>

typedef __attribute__((ext_vector_type(16))) _Float16 v16h;
typedef __attribute__((ext_vector_type(8)))  _Float16 v8h;
typedef __attribute__((ext_vector_type(8)))  float    v8f;
typedef __attribute__((ext_vector_type(4)))  unsigned int u32x4;
typedef __attribute__((ext_vector_type(4)))  int          i32x4;
typedef __attribute__((ext_vector_type(8)))  int          i32x8;

#define LDK 72   // padded row stride in halves (144B rows: 16B-aligned, bank-conflict pad)

// ---------------------------------------------------------------------------
// Tensor Data Mover: 2D tile load (rows x cols of 2-byte elems) into LDS,
// with TDM-applied LDS padding: pad_interval=32 DWORDs (128B), pad_amount=4
// DWORDs (16B)  ->  LDS row stride 144B == LDK halves.
// ---------------------------------------------------------------------------
__device__ __forceinline__ void tdm_load_tile_2d(unsigned ldsAddr, const void* gsrc,
                                                 unsigned tensorD0, unsigned tensorD1,
                                                 unsigned tileD0, unsigned tileD1,
                                                 unsigned strideD0 /*elements*/) {
    unsigned long long ga = (unsigned long long)(uintptr_t)gsrc;
    u32x4 g0;
    g0[0] = 1u;                                        // count=1 (valid user D#)
    g0[1] = ldsAddr;                                   // lds_addr (bytes)
    g0[2] = (unsigned)(ga & 0xffffffffu);              // global_addr[31:0]
    g0[3] = (unsigned)((ga >> 32) & 0x01ffffffu)       // global_addr[56:32]
          | (2u << 30);                                // type = 2 ("image")
    i32x8 g1;
    g1[0] = (1 << 16)                                  // data_size = 2 bytes
          | (1 << 20)                                  // pad_enable
          | (4 << 22)                                  // pad_interval: 32 DWORDs
          | (3 << 25);                                 // pad_amount: 4 DWORDs
    g1[1] = (int)((tensorD0 & 0xffffu) << 16);         // tensor_dim0[15:0]
    g1[2] = (int)((tensorD0 >> 16) | ((tensorD1 & 0xffffu) << 16));
    g1[3] = (int)((tensorD1 >> 16) | (tileD0 << 16));  // tile_dim0
    g1[4] = (int)(tileD1 & 0xffffu);                   // tile_dim1 (tile_dim2=0)
    g1[5] = (int)strideD0;                             // tensor_dim0_stride[31:0]
    g1[6] = 0;                                         // stride0[47:32], dim1_stride=0
    g1[7] = 0;
    i32x4 gz = {0, 0, 0, 0};
#if __clang_major__ >= 23
    i32x8 gz8 = {0, 0, 0, 0, 0, 0, 0, 0};
    __builtin_amdgcn_tensor_load_to_lds(g0, g1, gz, gz, gz8, 0);
#else
    __builtin_amdgcn_tensor_load_to_lds(g0, g1, gz, gz, 0);
#endif
}

// ---------------------------------------------------------------------------
// fp32 -> fp16 conversion with scale (weight gain folding)
// ---------------------------------------------------------------------------
__global__ __launch_bounds__(256)
void cvt_f32_f16(const float* __restrict__ in, _Float16* __restrict__ out,
                 float scale, long n) {
    long i = (long)blockIdx.x * blockDim.x + threadIdx.x;
    long step = (long)gridDim.x * blockDim.x;
    for (; i < n; i += step) out[i] = (_Float16)(in[i] * scale);
}

// ---------------------------------------------------------------------------
// depthwise 4x4 FIR ([1,3,3,1]/8 outer product), stride `down`, pad `pad`
// ---------------------------------------------------------------------------
template<int IH, int IW, int OH, int OW, int PAD, int DOWN>
__global__ __launch_bounds__(256)
void fir4x4(const _Float16* __restrict__ in, _Float16* __restrict__ out, int NC) {
    long idx = (long)blockIdx.x * blockDim.x + threadIdx.x;
    long total = (long)NC * OH * OW;
    if (idx >= total) return;
    int ow = (int)(idx % OW);
    long t = idx / OW;
    int oh = (int)(t % OH);
    int c  = (int)(t / OH);
    const float f[4] = {0.125f, 0.375f, 0.375f, 0.125f};
    const _Float16* base = in + (size_t)c * IH * IW;
    float acc = 0.f;
#pragma unroll
    for (int i = 0; i < 4; ++i) {
        int ih = oh * DOWN - PAD + i;
        if (ih < 0 || ih >= IH) continue;
#pragma unroll
        for (int j = 0; j < 4; ++j) {
            int iw = ow * DOWN - PAD + j;
            if (iw < 0 || iw >= IW) continue;
            acc += f[i] * f[j] * (float)base[ih * IW + iw];
        }
    }
    out[idx] = (_Float16)acc;
}

// ---------------------------------------------------------------------------
// Implicit-GEMM conv on v_wmma_f32_16x16x32_f16, all geometry compile-time.
//   M = Nb*OH*OW, Ndim = Cout, K = Cin*KH*KW
//   block = 256 threads = 8 waves (2 Mwaves x 4 Nwaves), tile 32M x 128N
//   K-step = 64 -> 4 WMMAs per wave per round
//   Software pipeline: double-buffered LDS; TDM (weights) + im2col stage for
//   step k+1 issued before the WMMAs of step k; one barrier per iteration.
//   epilogue: leaky_relu(x,0.2)*sqrt(2); optional fp32 accumulate
// ---------------------------------------------------------------------------
template<int Cin, int IH, int IW, int Cout, int OH, int OW,
         int KH, int KW, int STRIDE, int PAD, bool OUT_HALF, bool ACCUM>
__global__ __launch_bounds__(256)
void conv_wmma(const _Float16* __restrict__ in, const _Float16* __restrict__ wgt,
               _Float16* __restrict__ outH, float* __restrict__ outF, int Nb) {
    constexpr int Ktot = Cin * KH * KW;    // multiple of 64 for all configs
    constexpr int KHW  = KH * KW;
    __shared__ _Float16 lA[2][32 * LDK];   // ping-pong [Mrow][K0..63] im2col tiles
    __shared__ _Float16 lB[2][128 * LDK];  // ping-pong [Nloc][K0..63] weight tiles (TDM)

    const int tid   = threadIdx.x;
    const int wave  = tid >> 5;
    const int lane  = tid & 31;
    const int waveM = wave >> 2;           // 0..1
    const int waveN = wave & 3;            // 0..3
    const int hh    = lane >> 4;
    const int nl    = lane & 15;

    const int Mtot  = Nb * OH * OW;
    const int mBase = blockIdx.x * 32;
    const int nBase = blockIdx.y * 128;

    // ---- loop-invariant A-staging decode (8 halves per thread per round) ----
    const int rowA = tid >> 3;
    const int kk0A = (tid & 7) * 8;
    const int mA   = mBase + rowA;
    const bool mokA = (mA < Mtot);
    int nA = 0, ohA = 0, owA = 0;
    if (mokA) {
        nA = mA / (OH * OW);               // constant divisors -> mul/shift
        int r2 = mA - nA * (OH * OW);
        ohA = r2 / OW;
        owA = r2 - ohA * OW;
    }

    // stage one 32x64 im2col tile for K range [k0,k0+64) into dst
    auto stageA = [&](int k0, _Float16* dst) {
#pragma unroll
        for (int i = 0; i < 8; ++i) {
            int k = k0 + kk0A + i;
            _Float16 v = (_Float16)0.f;
            if (mokA) {
                int ci = k / KHW;
                int rr = k - ci * KHW;
                int kh = rr / KW;
                int kw = rr - kh * KW;
                int ih = ohA * STRIDE - PAD + kh;
                int iw = owA * STRIDE - PAD + kw;
                if (ih >= 0 && ih < IH && iw >= 0 && iw < IW) {
                    size_t gidx = (((size_t)nA * Cin + ci) * IH + ih) * IW + iw;
                    v = in[gidx];
                    if (i == 0 && k0 + 64 < Ktot)
                        __builtin_prefetch(in + gidx + (size_t)(64 / KHW) * IH * IW, 0, 1);
                }
            }
            dst[rowA * LDK + kk0A + i] = v;
        }
    };

    v8f acc0 = {}, acc1 = {};

    // ---- prologue: prime buffer 0 ----
    if (wave == 0) {
        tdm_load_tile_2d((unsigned)(uintptr_t)(&lB[0][0]),
                         wgt + (size_t)nBase * Ktot,
                         Ktot, Cout, 64, 128, Ktot);
    }
    stageA(0, &lA[0][0]);
    if (wave == 0) __builtin_amdgcn_s_wait_tensorcnt(0);
    __syncthreads();

    int p = 0;
    for (int k0 = 0; k0 < Ktot; k0 += 64) {
        const int kn = k0 + 64;
        const bool more = (kn < Ktot);

        // ---- issue next tile's staging before consuming current tile ----
        if (more) {
            if (wave == 0) {
                tdm_load_tile_2d((unsigned)(uintptr_t)(&lB[1 - p][0]),
                                 wgt + (size_t)nBase * Ktot + kn,
                                 Ktot, Cout, 64, 128, Ktot);
            }
            stageA(kn, &lA[1 - p][0]);
        }

        // ---- compute on buffer p: 2 K-substeps x 2 N-tiles = 4 WMMAs ----
        const _Float16* pa  = &lA[p][(waveM * 16 + nl) * LDK];
        const _Float16* pb0 = &lB[p][(waveN * 32 + nl) * LDK];
        const _Float16* pb1 = &lB[p][(waveN * 32 + 16 + nl) * LDK];
#pragma unroll
        for (int ks = 0; ks < 2; ++ks) {
            const int kf = ks * 32;
            // A 16x32 f16 layout: lanes0-15 K{0..7,16..23}; lanes16-31 K{8..15,24..31}
            v8h a0 = *reinterpret_cast<const v8h*>(pa + kf + hh * 8);
            v8h a1 = *reinterpret_cast<const v8h*>(pa + kf + 16 + hh * 8);
            // B 32x16 f16 layout: lanes0-15 K{0..7,16..23}; lanes16-31 K{8..15,24..31} (N=lane&15)
            v8h b0 = *reinterpret_cast<const v8h*>(pb0 + kf + hh * 8);
            v8h b1 = *reinterpret_cast<const v8h*>(pb0 + kf + 16 + hh * 8);
            v8h c0 = *reinterpret_cast<const v8h*>(pb1 + kf + hh * 8);
            v8h c1 = *reinterpret_cast<const v8h*>(pb1 + kf + 16 + hh * 8);
            v16h av, bv, cv;
#pragma unroll
            for (int i = 0; i < 8; ++i) {
                av[i] = a0[i]; av[8 + i] = a1[i];
                bv[i] = b0[i]; bv[8 + i] = b1[i];
                cv[i] = c0[i]; cv[8 + i] = c1[i];
            }
            acc0 = __builtin_amdgcn_wmma_f32_16x16x32_f16(false, av, false, bv,
                                                          (short)0, acc0, false, false);
            acc1 = __builtin_amdgcn_wmma_f32_16x16x32_f16(false, av, false, cv,
                                                          (short)0, acc1, false, false);
        }

        // ---- publish next buffer ----
        if (more) {
            if (wave == 0) __builtin_amdgcn_s_wait_tensorcnt(0);
            __syncthreads();
        }
        p ^= 1;
    }

    // ---- epilogue: fused leaky_relu * sqrt(2) ----
    const float gain = 1.4142135623730951f;
#pragma unroll
    for (int t = 0; t < 2; ++t) {
        v8f a = t ? acc1 : acc0;
        const int oc = nBase + waveN * 32 + t * 16 + nl;
#pragma unroll
        for (int r = 0; r < 8; ++r) {
            int m = mBase + waveM * 16 + (r + 8 * hh);  // C/D layout: M = r + 8*half
            if (m >= Mtot) continue;
            int n  = m / (OH * OW);
            int r2 = m - n * (OH * OW);
            int oh = r2 / OW;
            int ow = r2 - oh * OW;
            float v = a[r];
            v = (v > 0.f ? v : 0.2f * v) * gain;
            size_t oidx = (((size_t)n * Cout + oc) * OH + oh) * OW + ow;
            if (OUT_HALF) {
                outH[oidx] = (_Float16)v;
            } else {
                if (ACCUM) outF[oidx] += v;
                else       outF[oidx]  = v;
            }
        }
    }
}

// ---------------------------------------------------------------------------
extern "C" void kernel_launch(void* const* d_in, const int* in_sizes, int n_in,
                              void* d_out, int out_size, void* d_ws, size_t ws_size,
                              hipStream_t stream) {
    (void)in_sizes; (void)n_in; (void)out_size; (void)ws_size;

    const float* y  = (const float*)d_in[0];   // [16,256,64,64]
    const float* wc = (const float*)d_in[1];   // [256,256,3,3]
    const float* wd = (const float*)d_in[2];   // [512,256,3,3]
    const float* ws = (const float*)d_in[3];   // [512,256,1,1]
    float* out = (float*)d_out;                // [16,512,32,32]

    const int N = 16, Cin = 256, Cout = 512, H = 64, W = 64;
    const long nY  = (long)N * Cin * H * W;        // 16,777,216
    const long nX2 = (long)N * Cin * 65 * 65;      // 17,305,600
    const long nSk = (long)N * Cin * 32 * 32;      //  4,194,304
    const long nWc = (long)Cin * Cin * 9;          //    589,824
    const long nWd = (long)Cout * Cin * 9;         //  1,179,648
    const long nWs = (long)Cout * Cin;             //    131,072

    _Float16* ybf = (_Float16*)d_ws;
    _Float16* x1  = ybf + nY;
    _Float16* x2  = x1  + nY;
    _Float16* sk  = x2  + nX2;
    _Float16* wcb = sk  + nSk;
    _Float16* wdb = wcb + nWc;
    _Float16* wsb = wdb + nWd;

    const float g3 = 1.0f / 48.0f;   // 1/sqrt(256*9)
    const float g1 = 0.0625f;        // 1/sqrt(256)
    dim3 blk(256);

    cvt_f32_f16<<<4096, blk, 0, stream>>>(y,  ybf, 1.0f, nY);
    cvt_f32_f16<<<1024, blk, 0, stream>>>(wc, wcb, g3,   nWc);
    cvt_f32_f16<<<1024, blk, 0, stream>>>(wd, wdb, g3,   nWd);
    cvt_f32_f16<<< 512, blk, 0, stream>>>(ws, wsb, g1,   nWs);

    // conv1: 3x3 s1 p1, 256->256 @64x64, act -> x1 (fp16)
    {
        dim3 grid(65536 / 32, 256 / 128);   // (2048, 2)
        conv_wmma<256, 64, 64, 256, 64, 64, 3, 3, 1, 1, true, false>
            <<<grid, blk, 0, stream>>>(ybf, wcb, x1, nullptr, N);
    }
    // FIR pad=2 down=1 on x1 -> x2 [65x65]
    fir4x4<64, 64, 65, 65, 2, 1><<<(unsigned)((nX2 + 255) / 256), blk, 0, stream>>>(
        x1, x2, N * Cin);
    // FIR pad=1 down=2 on y -> sk [32x32]
    fir4x4<64, 64, 32, 32, 1, 2><<<(unsigned)((nSk + 255) / 256), blk, 0, stream>>>(
        ybf, sk, N * Cin);
    // skip: 1x1, 256->512 @32x32, act -> out (overwrite)
    {
        dim3 grid(16384 / 32, 512 / 128);   // (512, 4)
        conv_wmma<256, 32, 32, 512, 32, 32, 1, 1, 1, 0, false, false>
            <<<grid, blk, 0, stream>>>(sk, wsb, nullptr, out, N);
    }
    // down conv: 3x3 s2 p0 on x2 [65x65] -> 32x32, act, accumulate into out
    {
        dim3 grid(16384 / 32, 512 / 128);   // (512, 4)
        conv_wmma<256, 65, 65, 512, 32, 32, 3, 3, 2, 0, false, true>
            <<<grid, blk, 0, stream>>>(x2, wdb, nullptr, out, N);
    }
}
